// TokenEmbedder_86019605004407
// MI455X (gfx1250) — matrix operand order repaired and checked by reading, not compiled, not run on hardware
//
#include <hip/hip_runtime.h>
#include <cstdint>
#include <cstddef>

// ---------------------------------------------------------------------------
// MI455X (gfx1250) implementation.
// Dominant work: zx[i,j,z] = sum_c u[i,c] u[j,c] W3[c,z]  (17.2 GFLOP)
// restructured as a per-i GEMM, with the zj projection folded into B:
//   z[i] (512x128) = u (512x256) @ (diag(u_i) W3 + W2) (256x128)  (+ row terms)
// computed with V_WMMA_F32_16X16X32_BF16 (bf16 in, f32 accumulate): at the
// bf16 WMMA rate this is ~3.4us, under the 23.3 TB/s write roofline for the
// 134 MB z output (~5.8us) -> memory-bound, which f32 WMMA would not be.
// One block per i, looping over j-blocks, so W2+W3 (256KB) are read once per
// i; folding zj into the GEMM removes a 134MB per-element L2 gather.
// u j-tiles are staged LDS-ward with GLOBAL_LOAD_ASYNC_TO_LDS_B128
// (ASYNCcnt path) when the toolchain exposes the builtin.
// ---------------------------------------------------------------------------

typedef __attribute__((ext_vector_type(16))) __bf16 v16bf;
typedef __attribute__((ext_vector_type(8)))  float  v8f;
typedef int v4i __attribute__((__vector_size__(16)));

union FragBF { v16bf v; uint4 q[2]; };

__device__ __forceinline__ unsigned short f2bf(float f) {
  unsigned u = __builtin_bit_cast(unsigned, f);
  u += 0x7FFFu + ((u >> 16) & 1u);          // round-to-nearest-even
  return (unsigned short)(u >> 16);
}

#define L_       512
#define CS_      256
#define CZ_      128
// padded LDS row stride for 256 bf16 (512B) rows: +16B so lane m starts at
// bank (132*m)%64 = 4m -> 16 lanes cover all 64 banks, conflict-free b128.
#define ROWB_    528

#if defined(__AMDGCN__) && \
    __has_builtin(__builtin_amdgcn_global_load_async_to_lds_b128) && \
    __has_builtin(__builtin_amdgcn_s_wait_asynccnt)
#define HAVE_ASYNC_LDS 1
// toolchain signature: (v4i* global_src, v4i* lds_dst, imm offset, imm cpol)
#define GLD_ASYNC_B128(g, l)                                             \
  __builtin_amdgcn_global_load_async_to_lds_b128(                        \
      (v4i*)(g), (v4i*)(l), 0, 0)
#else
#define HAVE_ASYNC_LDS 0
#endif

// ---------------------------------------------------------------------------
// Kernel 1: s = E_aa[seq] + E_pos[res] + E_chain[chain]; LayerNorm -> u
// ---------------------------------------------------------------------------
__global__ void embed_ln_kernel(const int* __restrict__ seq,
                                const int* __restrict__ chain,
                                const int* __restrict__ res,
                                const float* __restrict__ E_aa,
                                const float* __restrict__ E_pos,
                                const float* __restrict__ E_chain,
                                const float* __restrict__ ln_w,
                                const float* __restrict__ ln_b,
                                float* __restrict__ s_out,
                                float* __restrict__ u_f32,
                                unsigned short* __restrict__ u_bf) {
  const int i = blockIdx.x;
  const int c = threadIdx.x;
  __shared__ float red[256];

  const int aa = seq[i], pi = res[i], ch = chain[i];
  float sv = E_aa[aa * CS_ + c] + E_pos[pi * CS_ + c] + E_chain[ch * CS_ + c];
  s_out[i * CS_ + c] = sv;

  red[c] = sv;
  __syncthreads();
  for (int off = 128; off > 0; off >>= 1) {
    if (c < off) red[c] += red[c + off];
    __syncthreads();
  }
  const float mu = red[0] * (1.0f / CS_);
  __syncthreads();
  const float d = sv - mu;
  red[c] = d * d;
  __syncthreads();
  for (int off = 128; off > 0; off >>= 1) {
    if (c < off) red[c] += red[c + off];
    __syncthreads();
  }
  const float var = red[0] * (1.0f / CS_);
  const float uv = d * rsqrtf(var + 1e-5f) * ln_w[c] + ln_b[c];
  u_f32[i * CS_ + c] = uv;
  u_bf[i * CS_ + c]  = f2bf(uv);
}

// ---------------------------------------------------------------------------
// Kernel 2: zi_b = u@W1 + b_pair   (zj is folded into the pair GEMM's B)
// ---------------------------------------------------------------------------
__global__ void proj_kernel(const float* __restrict__ u_f32,
                            const float* __restrict__ W_pair,
                            const float* __restrict__ b_pair,
                            float* __restrict__ zi_b) {
  const int i = blockIdx.x;
  const int z = threadIdx.x;
  __shared__ float ur[CS_];
  ur[z]       = u_f32[i * CS_ + z];
  ur[z + 128] = u_f32[i * CS_ + z + 128];
  __syncthreads();
  float a1 = 0.f;
  #pragma unroll 8
  for (int c = 0; c < CS_; ++c)
    a1 += ur[c] * W_pair[c * CZ_ + z];         // W1
  zi_b[i * CZ_ + z] = a1 + b_pair[z];
}

// ---------------------------------------------------------------------------
// Kernel 3: one block per i (512 blocks, 256 threads = 8 waves).
// Prologue: build WT = (diag(u_i) W3 + W2)^T in LDS once (packed b128 stores).
// Loop over 4 j-blocks: async-stage 128 bf16 u rows, 8 K-steps x 8 WMMA per
// wave, fused epilogue (zi+b, E_chainpair, E_relpos) and store.
// LDS ~137KB -> 2 blocks per 320KB WGP (4 waves/SIMD); amdgpu_waves_per_eu(4)
// raises the VGPR budget accordingly so nothing spills to scratch.
// ---------------------------------------------------------------------------
#define SM_WT    0
#define SM_UJ    (128 * ROWB_)                 // 67584
#define SM_UIF   (2 * 128 * ROWB_)             // 135168
#define SM_ZIR   (SM_UIF + CS_ * 4)            // 136192
#define SM_CHJ   (SM_ZIR + CZ_ * 4)            // 136704
#define SM_RSJ   (SM_CHJ + 128 * 4)            // 137216
#define SMEM_BYTES (SM_RSJ + 128 * 4)          // 137728

__global__ void __launch_bounds__(256)
__attribute__((amdgpu_waves_per_eu(4)))
pair_kernel(const int* __restrict__ chain,
            const int* __restrict__ res,
            const float* __restrict__ E_cp,
            const float* __restrict__ E_rp,
            const float* __restrict__ W_pair,
            const float* __restrict__ u_f32,
            const unsigned short* __restrict__ u_bf,
            const float* __restrict__ zi_b,
            float* __restrict__ z_out) {
  extern __shared__ unsigned char smem[];
  unsigned short* WT  = (unsigned short*)(smem + SM_WT);
  float*          UiF = (float*)(smem + SM_UIF);
  float*          ziR = (float*)(smem + SM_ZIR);
  int*            chJ = (int*)(smem + SM_CHJ);
  int*            rsJ = (int*)(smem + SM_RSJ);

  const int t = threadIdx.x;
  const int i = blockIdx.x;

  // ---- per-i prologue: u_i row + zi row ----
  UiF[t] = u_f32[i * CS_ + t];
  if (t < 128) ziR[t] = zi_b[i * CZ_ + t];
  __syncthreads();

  // ---- build WT[z][c] = bf16(u_i[c] * W3[c][z] + W2[c][z]);
  //      W2/W3 read z-coalesced, 8 channels packed per b128 LDS store ----
  {
    const int tz = t & 127, half = t >> 7;
    const float* W2 = W_pair + (size_t)CS_ * CZ_;
    const float* W3 = W_pair + (size_t)2 * CS_ * CZ_;
    unsigned short* wrow = WT + tz * (ROWB_ / 2);
    #pragma unroll 2
    for (int c0 = half * 128; c0 < half * 128 + 128; c0 += 8) {
      unsigned dw[4];
      #pragma unroll
      for (int q = 0; q < 4; ++q) {
        const int c = c0 + 2 * q;
        const float a = UiF[c]     * W3[(size_t)c * CZ_ + tz]       + W2[(size_t)c * CZ_ + tz];
        const float b = UiF[c + 1] * W3[(size_t)(c + 1) * CZ_ + tz] + W2[(size_t)(c + 1) * CZ_ + tz];
        dw[q] = (unsigned)f2bf(a) | ((unsigned)f2bf(b) << 16);
      }
      uint4 pk; pk.x = dw[0]; pk.y = dw[1]; pk.z = dw[2]; pk.w = dw[3];
      *(uint4*)(wrow + c0) = pk;
    }
  }

  const int lane = t & 31, w = t >> 5;
  const int wj = w & 3;                        // 32-row j quarter
  const int wz = w >> 2;                       // 64-col z half
  const int m = lane & 15, h = lane >> 4;
  const int ci = chain[i], ri = res[i];

  for (int jb = 0; jb < 4; ++jb) {
    const int j0 = jb * 128;
    __syncthreads();                           // LDS safe to overwrite

    if (t < 128) { chJ[t] = chain[j0 + t]; rsJ[t] = res[j0 + t]; }
    {
      const uint4* src = (const uint4*)u_bf;   // 32 uint4 per 256-ch row
      #pragma unroll
      for (int k = 0; k < 16; ++k) {
        const int idx = t * 16 + k;            // 0..4095
        const int row = idx >> 5, off = idx & 31;
        void* dst = smem + SM_UJ + row * ROWB_ + off * 16;
        const uint4* g = src + (size_t)(j0 + row) * 32 + off;
#if HAVE_ASYNC_LDS
        GLD_ASYNC_B128(g, dst);                // ASYNCcnt, no VGPR round-trip
#else
        *(uint4*)dst = *g;
#endif
      }
#if HAVE_ASYNC_LDS
      __builtin_amdgcn_s_wait_asynccnt(0);
#endif
    }
    __syncthreads();

    v8f acc[2][4];
    {
      v8f zero = {0.f, 0.f, 0.f, 0.f, 0.f, 0.f, 0.f, 0.f};
      #pragma unroll
      for (int a = 0; a < 2; ++a)
        #pragma unroll
        for (int b = 0; b < 4; ++b) acc[a][b] = zero;
    }

    // ---- K loop: 8 steps of 32 channels; 8 WMMA per step per wave ----
    #pragma unroll
    for (int ks = 0; ks < 8; ++ks) {
      const int c0 = ks * 32;
      FragBF A[2], B[4];
      #pragma unroll
      for (int jt = 0; jt < 2; ++jt) {         // A: 16x32 bf16, M=lane%16
        const int row = wj * 32 + jt * 16 + m;
        const unsigned char* base = smem + SM_UJ + row * ROWB_;
        A[jt].q[0] = *(const uint4*)(base + (c0 + 8 * h) * 2);
        A[jt].q[1] = *(const uint4*)(base + (c0 + 16 + 8 * h) * 2);
      }
      #pragma unroll
      for (int zt = 0; zt < 4; ++zt) {         // B: 32x16 bf16, N=lane%16
        const int zr = wz * 64 + zt * 16 + m;
        const unsigned char* base = smem + SM_WT + zr * ROWB_;
        B[zt].q[0] = *(const uint4*)(base + (c0 + 8 * h) * 2);
        B[zt].q[1] = *(const uint4*)(base + (c0 + 16 + 8 * h) * 2);
      }
      #pragma unroll
      for (int jt = 0; jt < 2; ++jt)
        #pragma unroll
        for (int zt = 0; zt < 4; ++zt)
          acc[jt][zt] = __builtin_amdgcn_wmma_f32_16x16x32_bf16(
              false, A[jt].v, false, B[zt].v, (short)0, acc[jt][zt], false, false);
    }

    // ---- fused epilogue: + (zi@W1 + b_pair) + E_chainpair + E_relpos ----
    #pragma unroll
    for (int jt = 0; jt < 2; ++jt) {
      #pragma unroll
      for (int zt = 0; zt < 4; ++zt) {
        const int zc = wz * 64 + zt * 16 + m;  // C layout: N = lane%16
        const float zi_v = ziR[zc];
        #pragma unroll
        for (int r = 0; r < 8; ++r) {          // C layout: M = r + 8*(lane/16)
          const int jl = wj * 32 + jt * 16 + h * 8 + r;
          const int j  = j0 + jl;
          const int cj = chJ[jl], rj = rsJ[jl];
          int dif = ri - rj;
          dif = dif < -32 ? -32 : (dif > 32 ? 32 : dif);
          const int rel = (ci == cj) ? (dif + 32) : 65;
          const int cp  = ci * 2 + cj;
          const float v = acc[jt][zt][r] + zi_v
                        + E_cp[cp * CZ_ + zc] + E_rp[rel * CZ_ + zc];
          z_out[((size_t)i * L_ + (size_t)j) * CZ_ + zc] = v;
        }
      }
    }
  }
}

// ---------------------------------------------------------------------------
extern "C" void kernel_launch(void* const* d_in, const int* in_sizes, int n_in,
                              void* d_out, int out_size, void* d_ws, size_t ws_size,
                              hipStream_t stream) {
  const int*   seq    = (const int*)d_in[0];
  const int*   chain  = (const int*)d_in[1];
  const int*   res    = (const int*)d_in[2];
  const float* E_aa   = (const float*)d_in[3];
  const float* E_pos  = (const float*)d_in[4];
  const float* E_chn  = (const float*)d_in[5];
  const float* E_cp   = (const float*)d_in[6];
  const float* E_rp   = (const float*)d_in[7];
  const float* W_pair = (const float*)d_in[8];
  const float* b_pair = (const float*)d_in[9];
  const float* ln_w   = (const float*)d_in[10];
  const float* ln_b   = (const float*)d_in[11];

  float* s_out = (float*)d_out;                       // 512*256
  float* z_out = (float*)d_out + (size_t)L_ * CS_;    // 512*512*128

  unsigned char* ws = (unsigned char*)d_ws;
  float*          u_f32 = (float*)(ws);                                  // 512KB
  unsigned short* u_bf  = (unsigned short*)(ws + (size_t)L_ * CS_ * 4);  // 256KB
  float*          zi_b  = (float*)(ws + (size_t)L_ * CS_ * 6);           // 256KB

  (void)in_sizes; (void)n_in; (void)out_size; (void)ws_size;

  (void)hipFuncSetAttribute((const void*)pair_kernel,
                            hipFuncAttributeMaxDynamicSharedMemorySize, SMEM_BYTES);

  embed_ln_kernel<<<L_, CS_, 0, stream>>>(seq, chain, res, E_aa, E_pos, E_chn,
                                          ln_w, ln_b, s_out, u_f32, u_bf);
  proj_kernel<<<L_, CZ_, 0, stream>>>(u_f32, W_pair, b_pair, zi_b);
  pair_kernel<<<L_, 256, SMEM_BYTES, stream>>>(
      chain, res, E_cp, E_rp, W_pair, u_f32, u_bf, zi_b, z_out);
}
